// Generator2_88553635709553
// MI455X (gfx1250) — compile-verified, matrix operand order checked by reading, not run on hardware
//
#include <hip/hip_runtime.h>
#include <cstdint>

// Problem constants (match reference)
#define B_  128
#define T_  256
#define C_  8
#define H_  1024
#define O_  2
#define KIN 32      // layer-0 input K padded 10 -> 32 for one WMMA k-step

#define NBLOCKS 64          // persistent grid: 64 WGP-resident workgroups
#define NTHREADS 256        // 8 waves -> 512 tile-waves total
#define HSZ ((size_t)B_ * H_)   // elements in one h buffer

typedef __attribute__((ext_vector_type(16))) __bf16 v16bf;
typedef __attribute__((ext_vector_type(8)))  __bf16 v8bf;
typedef __attribute__((ext_vector_type(8)))  float  v8f;
typedef __attribute__((ext_vector_type(4)))  float  v4f;

// ---------------------------------------------------------------------------
// Fragment loader (ISA 7.12.2, 16-bit 16x32 A layout; B mirrored with W's
// row-major (N,K) rows acting as B columns).
//   lane 0..15 : row/col = lane,    K = kBase + {0..7, 16..23}
//   lane 16..31: row/col = lane-16, K = kBase + {8..15, 24..31}
// Each half is a contiguous 16-byte chunk -> global_load_b128.
// ---------------------------------------------------------------------------
__device__ __forceinline__ v16bf load_frag(const __bf16* __restrict__ base,
                                           int rowStride, int rowBase,
                                           int kBase, int lane) {
    int r    = rowBase + (lane & 15);
    int kOff = kBase + ((lane & 16) ? 8 : 0);
    const __bf16* p = base + (size_t)r * rowStride + kOff;
    v8bf lo = *reinterpret_cast<const v8bf*>(p);
    v8bf hi = *reinterpret_cast<const v8bf*>(p + 16);
    return __builtin_shufflevector(lo, hi, 0,1,2,3,4,5,6,7,8,9,10,11,12,13,14,15);
}

// ---------------------------------------------------------------------------
// Device-scope grid barrier (generation counter). __threadfence() (agent
// seq_cst fence -> global_wb + global_inv on gfx1250) makes h-state written
// on other WGPs coherent past the per-CU L0. Relaxed volatile spin + s_sleep.
// Safe: gen can advance at most once past the value read here, because the
// next round needs this block's arrival too.
// ---------------------------------------------------------------------------
__device__ __forceinline__ void grid_sync(unsigned* cnt, volatile unsigned* gen,
                                          unsigned nb) {
    __syncthreads();
    if (threadIdx.x == 0) {
        __threadfence();                      // release my phase's writes
        unsigned g = *gen;
        unsigned prev = atomicAdd(cnt, 1u);
        if (prev == nb - 1u) {
            atomicExch(cnt, 0u);              // reset before release
            __threadfence();
            atomicAdd((unsigned*)gen, 1u);    // release everyone
        } else {
            while (*gen == g) __builtin_amdgcn_s_sleep(2);
        }
        __threadfence();                      // acquire others' writes
    }
    __syncthreads();
}

// ---------------------------------------------------------------------------
// One 16(batch) x 16(hidden) GRU-cell tile: 6 f32 accumulators (r,z,n for
// gi/gh), K-loops with v_wmma_f32_16x16x32_bf16, fused sigmoid/tanh blend.
// slot in [0,512): mTile = slot & 7, nTile = slot >> 3.
// All pointers are base-kernel-arg + offset so addrspace(1) is inferable
// (global_load_b128, not flat_load: avoids the DScnt coupling of FLAT ops).
// ---------------------------------------------------------------------------
__device__ __forceinline__ void gru_gates_tile(
        int slot, int lane,
        const __bf16* __restrict__ Xbf, int Kx,
        const __bf16* __restrict__ Wih, const __bf16* __restrict__ Whh,
        const float* __restrict__ bih, const float* __restrict__ bhh,
        const __bf16* __restrict__ Hbf, const float* __restrict__ Hf,
        __bf16* __restrict__ HbfOut, float* __restrict__ HfOut) {
    const int mBase = (slot & 7) * 16;
    const int nBase = (slot >> 3) * 16;

    v8f gi0 = {}, gi1 = {}, gi2 = {};
    v8f gh0 = {}, gh1 = {}, gh2 = {};

    // Recurrent GEMM: gh[g] += Hbf(16x32) * Whh_g(32x16), K = 1024
    for (int k = 0; k < H_; k += 32) {
        v16bf a  = load_frag(Hbf, H_, mBase, k, lane);
        v16bf br = load_frag(Whh, H_, 0 * H_ + nBase, k, lane);
        v16bf bz = load_frag(Whh, H_, 1 * H_ + nBase, k, lane);
        v16bf bn = load_frag(Whh, H_, 2 * H_ + nBase, k, lane);
        gh0 = __builtin_amdgcn_wmma_f32_16x16x32_bf16(false, a, false, br, (short)0, gh0, false, false);
        gh1 = __builtin_amdgcn_wmma_f32_16x16x32_bf16(false, a, false, bz, (short)0, gh1, false, false);
        gh2 = __builtin_amdgcn_wmma_f32_16x16x32_bf16(false, a, false, bn, (short)0, gh2, false, false);
    }
    // Input GEMM: gi[g] += Xbf(16x32) * Wih_g(32x16), K = Kx (32 or 1024)
    for (int k = 0; k < Kx; k += 32) {
        v16bf a  = load_frag(Xbf, Kx, mBase, k, lane);
        v16bf br = load_frag(Wih, Kx, 0 * H_ + nBase, k, lane);
        v16bf bz = load_frag(Wih, Kx, 1 * H_ + nBase, k, lane);
        v16bf bn = load_frag(Wih, Kx, 2 * H_ + nBase, k, lane);
        gi0 = __builtin_amdgcn_wmma_f32_16x16x32_bf16(false, a, false, br, (short)0, gi0, false, false);
        gi1 = __builtin_amdgcn_wmma_f32_16x16x32_bf16(false, a, false, bz, (short)0, gi1, false, false);
        gi2 = __builtin_amdgcn_wmma_f32_16x16x32_bf16(false, a, false, bn, (short)0, gi2, false, false);
    }

    // Epilogue. C/D layout: lane<16 -> rows mBase+0..7, lane>=16 -> +8.
    const int col = nBase + (lane & 15);
    const float bir = bih[col],          bhr = bhh[col];
    const float biz = bih[H_ + col],     bhz = bhh[H_ + col];
    const float bin = bih[2 * H_ + col], bhn = bhh[2 * H_ + col];
    const int rowOff = (lane & 16) ? 8 : 0;
#pragma unroll
    for (int i = 0; i < 8; ++i) {
        const int row = mBase + rowOff + i;
        float r = 1.f / (1.f + __expf(-(gi0[i] + bir + gh0[i] + bhr)));
        float z = 1.f / (1.f + __expf(-(gi1[i] + biz + gh1[i] + bhz)));
        float n = tanhf(gi2[i] + bin + r * (gh2[i] + bhn));
        float hOld = Hf[(size_t)row * H_ + col];
        float hNew = (1.f - z) * n + z * hOld;
        HfOut[(size_t)row * H_ + col]  = hNew;
        HbfOut[(size_t)row * H_ + col] = (__bf16)hNew;  // mirror feeds next GEMM
    }
}

// ---------------------------------------------------------------------------
// Persistent kernel: entire T=256 serial rollout in one launch.
// Per step: layer0 gates | barrier | layer1 gates | barrier | FC+feedback |
// barrier. h buffers are single 2x-sized allocations; ping-pong is an OFFSET
// (p * B*H) off the kernel-arg base so every access stays a GLOBAL load/store.
// ---------------------------------------------------------------------------
__global__ void __launch_bounds__(NTHREADS)
gru_persistent_kernel(const __bf16* __restrict__ wih0b, const __bf16* __restrict__ whh0b,
                      const __bf16* __restrict__ wih1b, const __bf16* __restrict__ whh1b,
                      const float* __restrict__ bih0, const float* __restrict__ bhh0,
                      const float* __restrict__ bih1, const float* __restrict__ bhh1,
                      const float* __restrict__ fcw,  const float* __restrict__ fcb,
                      const float* __restrict__ x,    float* __restrict__ out,
                      float*  __restrict__ h0f, float*  __restrict__ h1f,   // 2 x B x H
                      __bf16* __restrict__ h0b, __bf16* __restrict__ h1b,   // 2 x B x H
                      __bf16* __restrict__ xin,
                      unsigned* __restrict__ barCnt, unsigned* __restrict__ barGen) {
    const int lane = threadIdx.x & 31;
    const int wave = threadIdx.x >> 5;
    const int slot = blockIdx.x * 8 + wave;      // 0..511 tile-waves

    size_t offP = 0;                              // ping-pong offsets (elements)
    for (int t = 0; t < T_; ++t) {
        const size_t offQ = HSZ - offP;           // offP in {0, HSZ}

        // Layer 0: x_in (B x 32, K padded) + h0
        gru_gates_tile(slot, lane, xin, KIN, wih0b, whh0b, bih0, bhh0,
                       h0b + offP, h0f + offP, h0b + offQ, h0f + offQ);
        grid_sync(barCnt, barGen, gridDim.x);

        // Layer 1: h0_new (B x 1024) + h1
        gru_gates_tile(slot, lane, h0b + offQ, H_, wih1b, whh1b, bih1, bhh1,
                       h1b + offP, h1f + offP, h1b + offQ, h1f + offQ);
        grid_sync(barCnt, barGen, gridDim.x);

        // FC head + autoregressive feedback: one wave per (b, o) output.
        if (slot < B_ * O_) {
            const int b = slot >> 1, o = slot & 1;
            const v4f* hp = reinterpret_cast<const v4f*>(h1f + offQ + (size_t)b * H_);
            const v4f* wp = reinterpret_cast<const v4f*>(fcw + (size_t)o * H_);
            v4f a4 = {};
#pragma unroll
            for (int j = 0; j < 8; ++j)       // 8 x 32 lanes x float4 = 1024
                a4 += hp[j * 32 + lane] * wp[j * 32 + lane];
            float v = a4[0] + a4[1] + a4[2] + a4[3];
            for (int off = 16; off > 0; off >>= 1) v += __shfl_down(v, off);
            if (lane == 0) {
                float y = tanhf(v + fcb[o]);
                out[(size_t)b * T_ * O_ + (size_t)t * O_ + o] = y;
                xin[b * KIN + o] = (__bf16)y;
            }
            if (o == 0 && lane < C_)          // x[:, t, :] -> next input
                xin[b * KIN + O_ + lane] =
                    (__bf16)x[(size_t)b * T_ * C_ + (size_t)t * C_ + lane];
            // cols 10..31 stay zero (set once by init_kernel)
        }
        grid_sync(barCnt, barGen, gridDim.x);
        offP = offQ;
    }
}

// ---------------------------------------------------------------------------
// Setup kernels (run every call; d_ws is poisoned by the harness)
// ---------------------------------------------------------------------------
__global__ void f32_to_bf16_kernel(const float* __restrict__ src,
                                   __bf16* __restrict__ dst, int n) {
    int i = blockIdx.x * blockDim.x + threadIdx.x;
    if (i < n) dst[i] = (__bf16)src[i];
}

__global__ void pad_ih0_kernel(const float* __restrict__ src,  // (3H x 10)
                               __bf16* __restrict__ dst) {     // (3H x 32)
    int i = blockIdx.x * blockDim.x + threadIdx.x;             // 3H*32
    int row = i >> 5, col = i & 31;
    dst[i] = (col < 10) ? (__bf16)src[row * 10 + col] : (__bf16)0.f;
}

__global__ void init_kernel(const float* __restrict__ x,
                            const float* __restrict__ noise,
                            float* __restrict__ h0f, float* __restrict__ h1f,
                            __bf16* __restrict__ h0b, __bf16* __restrict__ h1b,
                            __bf16* __restrict__ xin,
                            unsigned* __restrict__ barCnt,
                            unsigned* __restrict__ barGen) {
    int i = blockIdx.x * blockDim.x + threadIdx.x;   // B*H (zeroes buffer 0)
    h0f[i] = 0.f; h1f[i] = 0.f;
    h0b[i] = (__bf16)0.f; h1b[i] = (__bf16)0.f;
    if (i == 0) { *barCnt = 0u; *barGen = 0u; }
    if (i < B_ * KIN) {
        int b = i >> 5, c = i & 31;
        float v = 0.f;
        if (c < O_) v = 1.f;                          // ones(B, O)
        else if (c < O_ + C_) {
            int ch = c - O_;
            // init_noise with last 4 channels replaced by x[:, 0, -4:]
            v = (ch < 4) ? noise[b * C_ + ch] : x[(size_t)b * T_ * C_ + ch];
        }
        xin[i] = (__bf16)v;
    }
}

// ---------------------------------------------------------------------------
extern "C" void kernel_launch(void* const* d_in, const int* in_sizes, int n_in,
                              void* d_out, int out_size, void* d_ws, size_t ws_size,
                              hipStream_t stream) {
    (void)in_sizes; (void)n_in; (void)out_size; (void)ws_size;
    const float* x          = (const float*)d_in[0];
    const float* init_noise = (const float*)d_in[1];
    /* d_in[2] = xlens, always T -> unused */
    const float* w_ih0 = (const float*)d_in[3];
    const float* w_hh0 = (const float*)d_in[4];
    const float* b_ih0 = (const float*)d_in[5];
    const float* b_hh0 = (const float*)d_in[6];
    const float* w_ih1 = (const float*)d_in[7];
    const float* w_hh1 = (const float*)d_in[8];
    const float* b_ih1 = (const float*)d_in[9];
    const float* b_hh1 = (const float*)d_in[10];
    const float* fc_w  = (const float*)d_in[11];
    const float* fc_b  = (const float*)d_in[12];
    float* out = (float*)d_out;

    // Workspace carve-out (~22 MB; whole working set is L2-resident)
    char* ws = (char*)d_ws;
    size_t off = 0;
    auto carve = [&](size_t bytes) {
        void* p = ws + off;
        off = (off + bytes + 255) & ~(size_t)255;
        return p;
    };
    __bf16* wih0b = (__bf16*)carve((size_t)3 * H_ * KIN * 2);
    __bf16* whh0b = (__bf16*)carve((size_t)3 * H_ * H_ * 2);
    __bf16* wih1b = (__bf16*)carve((size_t)3 * H_ * H_ * 2);
    __bf16* whh1b = (__bf16*)carve((size_t)3 * H_ * H_ * 2);
    float*  h0f = (float*)carve(2 * HSZ * 4);     // double-buffered
    float*  h1f = (float*)carve(2 * HSZ * 4);
    __bf16* h0b = (__bf16*)carve(2 * HSZ * 2);
    __bf16* h1b = (__bf16*)carve(2 * HSZ * 2);
    __bf16*   xin    = (__bf16*)carve((size_t)B_ * KIN * 2);
    unsigned* barCnt = (unsigned*)carve(256);
    unsigned* barGen = (unsigned*)carve(256);

    // Weight conversion to bf16
    const int NW = 3 * H_ * H_;
    f32_to_bf16_kernel<<<(NW + 255) / 256, 256, 0, stream>>>(w_hh0, whh0b, NW);
    f32_to_bf16_kernel<<<(NW + 255) / 256, 256, 0, stream>>>(w_ih1, wih1b, NW);
    f32_to_bf16_kernel<<<(NW + 255) / 256, 256, 0, stream>>>(w_hh1, whh1b, NW);
    pad_ih0_kernel<<<(3 * H_ * KIN + 255) / 256, 256, 0, stream>>>(w_ih0, wih0b);
    init_kernel<<<(B_ * H_) / 256, 256, 0, stream>>>(
        x, init_noise, h0f, h1f, h0b, h1b, xin, barCnt, barGen);

    // Entire serial rollout in a single persistent launch.
    gru_persistent_kernel<<<NBLOCKS, NTHREADS, 0, stream>>>(
        wih0b, whh0b, wih1b, whh1b,
        b_ih0, b_hh0, b_ih1, b_hh1,
        fc_w, fc_b, x, out,
        h0f, h1f, h0b, h1b,
        xin, barCnt, barGen);
}
